// AttentionModel_68779606278211
// MI455X (gfx1250) — compile-verified, matrix operand order checked by reading, not run on hardware
//
#include <hip/hip_runtime.h>
#include <hip/hip_bf16.h>
#include <math.h>

// ---------------- problem constants ----------------
#define Bn    128
#define Ln    200
#define Dn    128
#define Hn    8
#define DKn   16
#define FFn   512
#define NLn   2
#define NWn   3
#define BWn   384    // Bn*NWn
#define TOK   25600  // Bn*Ln
#define MTOK  76800  // BWn*Ln

typedef _Float16 f16;
typedef __attribute__((ext_vector_type(16))) _Float16 v16h;
typedef __attribute__((ext_vector_type(8)))  _Float16 v8h;
typedef __attribute__((ext_vector_type(8)))  float    v8f;

// =====================================================================
// Strided-batched GEMM:  C = alpha * A @ B (+bias) (+relu)
// A: [M,K] f32 row-major (lda); B: [K,N] (ldb), or transB -> stored [N,K]
// batch z = z0*inner + z1; per-operand offsets z0*s?0 + z1*s?1.
// f16 WMMA compute (v_wmma_f32_16x16x32_f16), f32 accumulate.
// Block = 128 threads = 4 waves. Wave w owns a 16x64 output tile:
// rows [blockIdx.x*64 + w*16), cols [blockIdx.y*64) (four 16-wide subtiles).
// CHECKED=false fast path requires M%64==0, N%64==0, K%32==0, transB==0.
// =====================================================================
template<bool CHECKED>
__global__ void k_gemm_wmma(
    const float* __restrict__ A, int lda, long sA0, long sA1,
    const float* __restrict__ Bm, int ldb, int transB, long sB0, long sB1,
    float* __restrict__ C, int ldc, long sC0, long sC1,
    int M, int N, int K, int inner,
    const float* __restrict__ bias, int relu, float alpha)
{
    __shared__ f16 As[4][16][40];   // [wave][m][k]
    __shared__ f16 Bs[64][40];      // [n][k] (transposed into LDS)

    int z  = blockIdx.z;
    int z0 = z / inner, z1 = z - z0 * inner;
    A  += (long)z0 * sA0 + (long)z1 * sA1;
    Bm += (long)z0 * sB0 + (long)z1 * sB1;
    C  += (long)z0 * sC0 + (long)z1 * sC1;

    const int w    = threadIdx.x >> 5;
    const int lane = threadIdx.x & 31;
    const int half = lane >> 4;
    const int mn   = lane & 15;
    const int tileM = blockIdx.x * 64 + w * 16;
    const int n0    = blockIdx.y * 64;

    v8f acc[4];
    #pragma unroll
    for (int s = 0; s < 4; ++s)
        acc[s] = (v8f){0.f, 0.f, 0.f, 0.f, 0.f, 0.f, 0.f, 0.f};

    for (int k0 = 0; k0 < K; k0 += 32) {
        if (!CHECKED) {
            // ---- fast branchless staging, vectorized ----
            // A: lane covers row r = lane>>1, k half kh = lane&1 (16 floats)
            {
                int r = lane >> 1, kh = lane & 1;
                const float* ap = A + (long)(tileM + r) * lda + k0 + kh * 16;
                float4 f0 = *(const float4*)(ap + 0);
                float4 f1 = *(const float4*)(ap + 4);
                float4 f2 = *(const float4*)(ap + 8);
                float4 f3 = *(const float4*)(ap + 12);
                if (k0 + 32 < K) __builtin_prefetch(ap + 32, 0, 3);
                v8h p0, p1;
                p0[0] = (f16)f0.x; p0[1] = (f16)f0.y; p0[2] = (f16)f0.z; p0[3] = (f16)f0.w;
                p0[4] = (f16)f1.x; p0[5] = (f16)f1.y; p0[6] = (f16)f1.z; p0[7] = (f16)f1.w;
                p1[0] = (f16)f2.x; p1[1] = (f16)f2.y; p1[2] = (f16)f2.z; p1[3] = (f16)f2.w;
                p1[4] = (f16)f3.x; p1[5] = (f16)f3.y; p1[6] = (f16)f3.z; p1[7] = (f16)f3.w;
                *(v8h*)&As[w][r][kh * 16 + 0] = p0;
                *(v8h*)&As[w][r][kh * 16 + 8] = p1;
            }
            // B: thread covers k row r2 = tid>>2, 16 n values q*16.. (transposed store)
            {
                int r2 = threadIdx.x >> 2, q = threadIdx.x & 3;
                const float* bp = Bm + (long)(k0 + r2) * ldb + n0 + q * 16;
                float4 g0 = *(const float4*)(bp + 0);
                float4 g1 = *(const float4*)(bp + 4);
                float4 g2 = *(const float4*)(bp + 8);
                float4 g3 = *(const float4*)(bp + 12);
                if (k0 + 32 < K) __builtin_prefetch(bp + (long)32 * ldb, 0, 3);
                f16 bh[16];
                bh[0]  = (f16)g0.x; bh[1]  = (f16)g0.y; bh[2]  = (f16)g0.z; bh[3]  = (f16)g0.w;
                bh[4]  = (f16)g1.x; bh[5]  = (f16)g1.y; bh[6]  = (f16)g1.z; bh[7]  = (f16)g1.w;
                bh[8]  = (f16)g2.x; bh[9]  = (f16)g2.y; bh[10] = (f16)g2.z; bh[11] = (f16)g2.w;
                bh[12] = (f16)g3.x; bh[13] = (f16)g3.y; bh[14] = (f16)g3.z; bh[15] = (f16)g3.w;
                #pragma unroll
                for (int j = 0; j < 16; ++j) Bs[q * 16 + j][r2] = bh[j];
            }
        } else {
            // ---- checked scalar staging (attention-shaped GEMMs) ----
            #pragma unroll
            for (int i = 0; i < 16; ++i) {
                int idx = lane * 16 + i;
                int r = idx >> 5, c = idx & 31;
                int m = tileM + r, k = k0 + c;
                As[w][r][c] = (m < M && k < K) ? (f16)A[(long)m * lda + k] : (f16)0.f;
            }
            for (int idx = threadIdx.x; idx < 2048; idx += 128) {
                int n = idx >> 5, c = idx & 31;
                f16 v = (f16)0.f;
                if (n0 + n < N && k0 + c < K)
                    v = (f16)(transB ? Bm[(long)(n0 + n) * ldb + (k0 + c)]
                                     : Bm[(long)(k0 + c) * ldb + (n0 + n)]);
                Bs[n][c] = v;
            }
        }
        __syncthreads();

        // fragment loads per ISA 16-bit layouts:
        // element i of lane -> k = (i&7) + 8*half + 16*(i>>3)  (two b128 runs)
        v8h alo = *(const v8h*)&As[w][mn][8 * half];
        v8h ahi = *(const v8h*)&As[w][mn][16 + 8 * half];
        v16h a;
        #pragma unroll
        for (int i = 0; i < 8; ++i) { a[i] = alo[i]; a[i + 8] = ahi[i]; }
        #pragma unroll
        for (int s = 0; s < 4; ++s) {
            v8h blo = *(const v8h*)&Bs[s * 16 + mn][8 * half];
            v8h bhi = *(const v8h*)&Bs[s * 16 + mn][16 + 8 * half];
            v16h b;
            #pragma unroll
            for (int i = 0; i < 8; ++i) { b[i] = blo[i]; b[i + 8] = bhi[i]; }
            acc[s] = __builtin_amdgcn_wmma_f32_16x16x32_f16(
                false, a, false, b, (short)0, acc[s], false, false);
        }
        __syncthreads();
    }

    // 32-bit C/D layout: lane holds col n = lane&15, rows m = j + 8*half
    #pragma unroll
    for (int j = 0; j < 8; ++j) {
        int m = tileM + j + 8 * half;
        #pragma unroll
        for (int s = 0; s < 4; ++s) {
            int n = n0 + s * 16 + mn;
            if (CHECKED && (m >= M || n >= N)) continue;
            float v = acc[s][j] * alpha;
            if (bias) v += bias[n];
            if (relu) v = fmaxf(v, 0.f);
            C[(long)m * ldc + n] = v;
        }
    }
}

// =====================================================================
// small elementwise / reduction kernels
// =====================================================================
__global__ void k_init_embed(const float* __restrict__ in,
                             const float* __restrict__ W,
                             const float* __restrict__ b,
                             float* __restrict__ h)
{
    long i = blockIdx.x * (long)blockDim.x + threadIdx.x;
    if (i >= (long)TOK * Dn) return;
    long m = i >> 7; int d = (int)(i & 127);
    float s = b[d];
    #pragma unroll
    for (int c = 0; c < 4; ++c) s += in[m * 4 + c] * W[c * Dn + d];
    h[i] = s;
}

// repack enc_Wq/Wk/Wv [H,D,DK] -> [D, H*DK]
__global__ void k_repack(const float* __restrict__ src, float* __restrict__ dst)
{
    int i = blockIdx.x * blockDim.x + threadIdx.x;
    if (i >= Dn * Dn) return;
    int d = i >> 7, c = i & 127, hh = c >> 4, kk = c & 15;
    dst[i] = src[(hh * Dn + d) * DKn + kk];
}

__global__ void k_softmax_rows(float* __restrict__ S, long rows, int cols)
{
    long r = blockIdx.x * (long)blockDim.x + threadIdx.x;
    if (r >= rows) return;
    float* p = S + r * cols;
    float m = -1e30f;
    for (int c = 0; c < cols; ++c) m = fmaxf(m, p[c]);
    float s = 0.f;
    for (int c = 0; c < cols; ++c) { float e = __expf(p[c] - m); p[c] = e; s += e; }
    float inv = 1.f / s;
    for (int c = 0; c < cols; ++c) p[c] *= inv;
}

__global__ void k_add(const float* __restrict__ a, const float* __restrict__ b,
                      float* __restrict__ c, long n)
{
    long i = blockIdx.x * (long)blockDim.x + threadIdx.x;
    if (i < n) c[i] = a[i] + b[i];
}

__global__ void k_zero(float* __restrict__ p, int n)
{
    int i = blockIdx.x * blockDim.x + threadIdx.x;
    if (i < n) p[i] = 0.f;
}

// sum / sumsq per channel (d = i & 127) with LDS pre-reduction
__global__ void k_bn_stats(const float* __restrict__ x, float* __restrict__ stats, long n)
{
    __shared__ float ls[Dn], lq[Dn];
    int tid = threadIdx.x;
    if (tid < Dn) { ls[tid] = 0.f; lq[tid] = 0.f; }
    __syncthreads();
    for (long i = blockIdx.x * (long)blockDim.x + tid; i < n;
         i += (long)gridDim.x * blockDim.x) {
        float v = x[i];
        int d = (int)(i & 127);
        atomicAdd(&ls[d], v);
        atomicAdd(&lq[d], v * v);
    }
    __syncthreads();
    if (tid < Dn) {
        atomicAdd(&stats[tid], ls[tid]);
        atomicAdd(&stats[Dn + tid], lq[tid]);
    }
}

__global__ void k_bn_apply(const float* __restrict__ x, const float* __restrict__ stats,
                           const float* __restrict__ g, const float* __restrict__ b,
                           float* __restrict__ out, long n, float invcnt)
{
    long i = blockIdx.x * (long)blockDim.x + threadIdx.x;
    if (i >= n) return;
    int d = (int)(i & 127);
    float mean = stats[d] * invcnt;
    float var  = stats[Dn + d] * invcnt - mean * mean;
    out[i] = (x[i] - mean) * rsqrtf(var + 1e-5f) * g[d] + b[d];
}

// RoutingNet: coef[w] = tanh((relu(w_cat @ embW + embB) @ outW)[:,0] + outB[0] + w_cat[:,0])
__global__ void k_routing(const float* __restrict__ w,
                          const float* __restrict__ embW, const float* __restrict__ embB,
                          const float* __restrict__ outW, const float* __restrict__ outB,
                          float* __restrict__ coef)
{
    __shared__ float red[NWn][Dn];
    int d = threadIdx.x; // 128 threads
    float wc[NWn][3];
    #pragma unroll
    for (int i = 0; i < NWn; ++i) {
        wc[i][0] = w[i * 2 + 0];
        wc[i][1] = w[i * 2 + 1];
        wc[i][2] = 0.f;
    }
    #pragma unroll
    for (int i = 0; i < NWn; ++i) {
        float e = embB[d];
        #pragma unroll
        for (int c = 0; c < 3; ++c) e += wc[i][c] * embW[c * Dn + d];
        e = fmaxf(e, 0.f);
        red[i][d] = e * outW[d * 3 + 0];
    }
    __syncthreads();
    for (int s = 64; s > 0; s >>= 1) {
        if (d < s)
            for (int i = 0; i < NWn; ++i) red[i][d] += red[i][d + s];
        __syncthreads();
    }
    if (d < NWn) coef[d] = tanhf(red[d][0] + outB[0] + wc[d][0]);
}

__global__ void k_mix(const float* __restrict__ h, const float* __restrict__ coef,
                      float* __restrict__ mixed, long n)
{
    long i = blockIdx.x * (long)blockDim.x + threadIdx.x;
    if (i >= n) return;
    long bw = i / (Ln * Dn);
    long rest = i - bw * (Ln * Dn);
    long b = bw / NWn;
    int wv = (int)(bw - b * NWn);
    mixed[i] = coef[wv] * h[b * (Ln * Dn) + rest];
}

__global__ void k_mean(const float* __restrict__ mixed, float* __restrict__ mm)
{
    int i = blockIdx.x * blockDim.x + threadIdx.x;
    if (i >= BWn * Dn) return;
    int bw = i >> 7, d = i & 127;
    const float* p = mixed + (long)bw * (Ln * Dn) + d;
    float s = 0.f;
    for (int l = 0; l < Ln; ++l) s += p[l * Dn];
    mm[i] = s * (1.f / (float)Ln);
}

// =====================================================================
// autoregressive greedy decode: one block per BW row, 200 steps inside.
// gall row layout: [0:128)=gK(h*16+kk), [128:256)=gV, [256:384)=logK
// =====================================================================
__global__ void k_decode(const float* __restrict__ mixed,
                         const float* __restrict__ fctx,
                         const float* __restrict__ gall,
                         const float* __restrict__ Wstep,
                         const float* __restrict__ Wout,
                         float* __restrict__ out_logp,
                         float* __restrict__ out_pis)
{
    __shared__ float q[Dn], prevrow[Dn], heads[Dn], glimpse[Dn];
    __shared__ float compat[Hn * Ln];
    __shared__ float logits[Ln];
    __shared__ float redv[256];
    __shared__ int   redi[256];
    __shared__ unsigned char visited[Ln];
    __shared__ int prevS;

    const int bw  = blockIdx.x;
    const int tid = threadIdx.x;
    const float* mix = mixed + (long)bw * Ln * Dn;
    const float* fc  = fctx + bw * Dn;
    const float* ga  = gall + (long)bw * Ln * 384;
    const float inv_sqrt_dk = 0.25f;
    const float inv_sqrt_d  = 0.08838834764831845f; // 1/sqrt(128)

    for (int n = tid; n < Ln; n += 256) visited[n] = 0;
    if (tid == 0) prevS = 0;
    __syncthreads();

    for (int t = 0; t < Ln; ++t) {
        int prev = prevS;
        if (tid < Dn) prevrow[tid] = mix[(long)prev * Dn + tid];
        __syncthreads();

        // query = fixed_ctx + mixed[prev] @ Wstep
        if (tid < Dn) {
            float s = fc[tid];
            for (int e = 0; e < Dn; ++e) s += prevrow[e] * Wstep[e * Dn + tid];
            q[tid] = s;
        }
        __syncthreads();

        // compat[h][n] = (q_h . gK[n,h]) / sqrt(DK), masked
        for (int idx = tid; idx < Hn * Ln; idx += 256) {
            int hh = idx / Ln, n = idx - hh * Ln;
            const float* gk = ga + (long)n * 384 + hh * DKn;
            float s = 0.f;
            #pragma unroll
            for (int kk = 0; kk < DKn; ++kk) s += q[hh * DKn + kk] * gk[kk];
            s *= inv_sqrt_dk;
            compat[hh * Ln + n] = visited[n] ? -1e9f : s;
        }
        __syncthreads();

        // per-head softmax + weighted sum of gV -> heads[h*16+kk]
        {
            int hh = tid >> 5, lane = tid & 31;
            float m = -1e30f;
            for (int n = lane; n < Ln; n += 32) m = fmaxf(m, compat[hh * Ln + n]);
            for (int off = 16; off > 0; off >>= 1) m = fmaxf(m, __shfl_xor(m, off, 32));
            float s = 0.f;
            for (int n = lane; n < Ln; n += 32) {
                float e = __expf(compat[hh * Ln + n] - m);
                compat[hh * Ln + n] = e;
                s += e;
            }
            for (int off = 16; off > 0; off >>= 1) s += __shfl_xor(s, off, 32);
            float inv = 1.f / s;
            float part[DKn];
            #pragma unroll
            for (int kk = 0; kk < DKn; ++kk) part[kk] = 0.f;
            for (int n = lane; n < Ln; n += 32) {
                float a = compat[hh * Ln + n];
                const float* gv = ga + (long)n * 384 + Dn + hh * DKn;
                #pragma unroll
                for (int kk = 0; kk < DKn; ++kk) part[kk] += a * gv[kk];
            }
            #pragma unroll
            for (int kk = 0; kk < DKn; ++kk) {
                float v = part[kk];
                for (int off = 16; off > 0; off >>= 1) v += __shfl_xor(v, off, 32);
                if (lane == 0) heads[hh * DKn + kk] = v * inv;
            }
        }
        __syncthreads();

        // glimpse = heads @ Wout
        if (tid < Dn) {
            float s = 0.f;
            for (int e = 0; e < Dn; ++e) s += heads[e] * Wout[e * Dn + tid];
            glimpse[tid] = s;
        }
        __syncthreads();

        // logits[n] = clip*tanh(glimpse . logK[n] / sqrt(D)), masked
        for (int n = tid; n < Ln; n += 256) {
            const float* lk = ga + (long)n * 384 + 256;
            float s = 0.f;
            for (int d = 0; d < Dn; ++d) s += glimpse[d] * lk[d];
            float lg = tanhf(s * inv_sqrt_d) * 10.f;
            logits[n] = visited[n] ? -1e9f : lg;
        }
        __syncthreads();

        // block argmax (first index on ties)
        float bm = -1e30f; int bi = 0;
        for (int n = tid; n < Ln; n += 256) {
            float v = logits[n];
            if (v > bm || (v == bm && n < bi)) { bm = v; bi = n; }
        }
        redv[tid] = bm; redi[tid] = bi;
        __syncthreads();
        for (int s = 128; s > 0; s >>= 1) {
            if (tid < s) {
                float v = redv[tid + s]; int ix = redi[tid + s];
                if (v > redv[tid] || (v == redv[tid] && ix < redi[tid])) {
                    redv[tid] = v; redi[tid] = ix;
                }
            }
            __syncthreads();
        }
        float mx = redv[0];
        int sel  = redi[0];
        __syncthreads();

        // log-sum-exp
        float ps = 0.f;
        for (int n = tid; n < Ln; n += 256) ps += __expf(logits[n] - mx);
        redv[tid] = ps;
        __syncthreads();
        for (int s = 128; s > 0; s >>= 1) {
            if (tid < s) redv[tid] += redv[tid + s];
            __syncthreads();
        }
        float lse = mx + logf(redv[0]);
        __syncthreads();

        float* outrow = out_logp + ((long)bw * Ln + t) * Ln;
        for (int n = tid; n < Ln; n += 256) outrow[n] = logits[n] - lse;
        if (tid == 0) {
            visited[sel] = 1;
            prevS = sel;
            out_pis[(long)bw * Ln + t] = (float)sel;
        }
        __syncthreads();
    }
}

// =====================================================================
// host-side orchestration
// =====================================================================
extern "C" void kernel_launch(void* const* d_in, const int* in_sizes, int n_in,
                              void* d_out, int out_size, void* d_ws, size_t ws_size,
                              hipStream_t stream)
{
    const float* input    = (const float*)d_in[0];
    const float* w_in     = (const float*)d_in[1];
    const float* init_W   = (const float*)d_in[2];
    const float* init_b   = (const float*)d_in[3];
    const float* enc_Wq   = (const float*)d_in[4];
    const float* enc_Wk   = (const float*)d_in[5];
    const float* enc_Wv   = (const float*)d_in[6];
    const float* enc_Wo   = (const float*)d_in[7];
    const float* bn1_g    = (const float*)d_in[8];
    const float* bn1_b    = (const float*)d_in[9];
    const float* ff1_W    = (const float*)d_in[10];
    const float* ff1_b    = (const float*)d_in[11];
    const float* ff2_W    = (const float*)d_in[12];
    const float* ff2_b    = (const float*)d_in[13];
    const float* bn2_g    = (const float*)d_in[14];
    const float* bn2_b    = (const float*)d_in[15];
    const float* rn_emb_W = (const float*)d_in[16];
    const float* rn_emb_b = (const float*)d_in[17];
    const float* rn_out_W = (const float*)d_in[18];
    const float* rn_out_b = (const float*)d_in[19];
    const float* Wnode    = (const float*)d_in[20];
    const float* Wfixed   = (const float*)d_in[21];
    const float* Wstep    = (const float*)d_in[22];
    const float* Wout     = (const float*)d_in[23];

    float* ws = (float*)d_ws;
    size_t off = 0;
    auto alloc = [&](size_t n) { float* p = ws + off; off += n; return p; };

    float* hbuf   = alloc((size_t)TOK * Dn);
    float* xbuf   = alloc((size_t)TOK * Dn);
    float* qb     = alloc((size_t)TOK * Dn);
    float* kb     = alloc((size_t)TOK * Dn);
    float* vb     = alloc((size_t)TOK * Dn);
    float* Wqp    = alloc((size_t)Dn * Dn);
    float* Wkp    = alloc((size_t)Dn * Dn);
    float* Wvp    = alloc((size_t)Dn * Dn);
    float* scores = alloc((size_t)Bn * Hn * Ln * Ln);
    float* attout = alloc((size_t)TOK * Dn);
    float* mhbuf  = alloc((size_t)TOK * Dn);
    float* ffb    = alloc((size_t)TOK * FFn);
    float* ff2b   = alloc((size_t)TOK * Dn);
    float* stats  = alloc(2 * Dn);
    float* coef   = alloc(4);
    float* mixedb = alloc((size_t)MTOK * Dn);
    float* mmean  = alloc((size_t)BWn * Dn);
    float* fctxb  = alloc((size_t)BWn * Dn);
    float* gallb  = alloc((size_t)MTOK * 384);
    (void)ws_size; (void)n_in; (void)in_sizes; (void)out_size;

    auto gemm = [&](const float* A, int lda, long sA0, long sA1,
                    const float* B, int ldb, int transB, long sB0, long sB1,
                    float* C, int ldc, long sC0, long sC1,
                    int M, int N, int K, int batch, int inner,
                    const float* bias, int relu, float alpha) {
        dim3 grid((M + 63) / 64, (N + 63) / 64, batch);
        bool fast = ((M & 63) == 0) && ((N & 63) == 0) && ((K & 31) == 0) && !transB;
        if (fast)
            k_gemm_wmma<false><<<grid, dim3(128), 0, stream>>>(
                A, lda, sA0, sA1, B, ldb, transB, sB0, sB1,
                C, ldc, sC0, sC1, M, N, K, inner, bias, relu, alpha);
        else
            k_gemm_wmma<true><<<grid, dim3(128), 0, stream>>>(
                A, lda, sA0, sA1, B, ldb, transB, sB0, sB1,
                C, ldc, sC0, sC1, M, N, K, inner, bias, relu, alpha);
    };

    const long nTok = (long)TOK * Dn;

    // ---- init embed ----
    k_init_embed<<<(int)((nTok + 255) / 256), 256, 0, stream>>>(input, init_W, init_b, hbuf);

    // ---- encoder layers ----
    for (int l = 0; l < NLn; ++l) {
        const float* Wq = enc_Wq + (size_t)l * Hn * Dn * DKn;
        const float* Wk = enc_Wk + (size_t)l * Hn * Dn * DKn;
        const float* Wv = enc_Wv + (size_t)l * Hn * Dn * DKn;
        const float* Wo = enc_Wo + (size_t)l * Hn * DKn * Dn; // already [H*DK, D] flat

        k_repack<<<64, 256, 0, stream>>>(Wq, Wqp);
        k_repack<<<64, 256, 0, stream>>>(Wk, Wkp);
        k_repack<<<64, 256, 0, stream>>>(Wv, Wvp);

        gemm(hbuf, Dn, 0, 0, Wqp, Dn, 0, 0, 0, qb, Dn, 0, 0, TOK, Dn, Dn, 1, 1, nullptr, 0, 1.f);
        gemm(hbuf, Dn, 0, 0, Wkp, Dn, 0, 0, 0, kb, Dn, 0, 0, TOK, Dn, Dn, 1, 1, nullptr, 0, 1.f);
        gemm(hbuf, Dn, 0, 0, Wvp, Dn, 0, 0, 0, vb, Dn, 0, 0, TOK, Dn, Dn, 1, 1, nullptr, 0, 1.f);

        // scores[b,h] = q[b,:,h,:] @ k[b,:,h,:]^T / sqrt(DK)
        gemm(qb, Dn, (long)Ln * Dn, DKn,
             kb, Dn, 1, (long)Ln * Dn, DKn,
             scores, Ln, (long)Hn * Ln * Ln, (long)Ln * Ln,
             Ln, Ln, DKn, Bn * Hn, Hn, nullptr, 0, 0.25f);

        k_softmax_rows<<<(Bn * Hn * Ln + 255) / 256, 256, 0, stream>>>(
            scores, (long)Bn * Hn * Ln, Ln);

        // attout[b,:,h,:] = softmax(scores) @ v[b,:,h,:]
        gemm(scores, Ln, (long)Hn * Ln * Ln, (long)Ln * Ln,
             vb, Dn, 0, (long)Ln * Dn, DKn,
             attout, Dn, (long)Ln * Dn, DKn,
             Ln, DKn, Ln, Bn * Hn, Hn, nullptr, 0, 1.f);

        // mh = attout @ Wo_flat
        gemm(attout, Dn, 0, 0, Wo, Dn, 0, 0, 0, mhbuf, Dn, 0, 0, TOK, Dn, Dn, 1, 1, nullptr, 0, 1.f);

        // BN1(h + mh)
        k_add<<<(int)((nTok + 255) / 256), 256, 0, stream>>>(hbuf, mhbuf, xbuf, nTok);
        k_zero<<<1, 256, 0, stream>>>(stats, 2 * Dn);
        k_bn_stats<<<512, 256, 0, stream>>>(xbuf, stats, nTok);
        k_bn_apply<<<(int)((nTok + 255) / 256), 256, 0, stream>>>(
            xbuf, stats, bn1_g + l * Dn, bn1_b + l * Dn, hbuf, nTok, 1.f / (float)TOK);

        // FF
        gemm(hbuf, Dn, 0, 0, ff1_W + (size_t)l * Dn * FFn, FFn, 0, 0, 0,
             ffb, FFn, 0, 0, TOK, FFn, Dn, 1, 1, ff1_b + l * FFn, 1, 1.f);
        gemm(ffb, FFn, 0, 0, ff2_W + (size_t)l * FFn * Dn, Dn, 0, 0, 0,
             ff2b, Dn, 0, 0, TOK, Dn, FFn, 1, 1, ff2_b + l * Dn, 0, 1.f);

        // BN2(h + ff)
        k_add<<<(int)((nTok + 255) / 256), 256, 0, stream>>>(hbuf, ff2b, xbuf, nTok);
        k_zero<<<1, 256, 0, stream>>>(stats, 2 * Dn);
        k_bn_stats<<<512, 256, 0, stream>>>(xbuf, stats, nTok);
        k_bn_apply<<<(int)((nTok + 255) / 256), 256, 0, stream>>>(
            xbuf, stats, bn2_g + l * Dn, bn2_b + l * Dn, hbuf, nTok, 1.f / (float)TOK);
    }

    // ---- routing + mixing ----
    k_routing<<<1, 128, 0, stream>>>(w_in, rn_emb_W, rn_emb_b, rn_out_W, rn_out_b, coef);
    const long nMix = (long)MTOK * Dn;
    k_mix<<<(int)((nMix + 255) / 256), 256, 0, stream>>>(hbuf, coef, mixedb, nMix);

    // ---- precompute ----
    k_mean<<<(BWn * Dn + 255) / 256, 256, 0, stream>>>(mixedb, mmean);
    gemm(mmean, Dn, 0, 0, Wfixed, Dn, 0, 0, 0, fctxb, Dn, 0, 0, BWn, Dn, Dn, 1, 1, nullptr, 0, 1.f);
    gemm(mixedb, Dn, 0, 0, Wnode, 3 * Dn, 0, 0, 0, gallb, 3 * Dn, 0, 0,
         MTOK, 3 * Dn, Dn, 1, 1, nullptr, 0, 1.f);

    // ---- decode ----
    float* out_logp = (float*)d_out;
    float* out_pis  = out_logp + (size_t)BWn * Ln * Ln;
    k_decode<<<BWn, 256, 0, stream>>>(mixedb, fctxb, gallb, Wstep, Wout, out_logp, out_pis);
}
